// Feedforward_61074434949468
// MI455X (gfx1250) — matrix operand order repaired and checked by reading, not compiled, not run on hardware
//
#include <hip/hip_runtime.h>
#include <hip/hip_bf16.h>

// ---------------------------------------------------------------------------
// CDNA5 / gfx1250: 3-layer [L2norm -> Linear -> ReLU] MLP, fp32 in/out.
// GEMMs run on the bf16 WMMA pipe using the split-fp32 (hi+lo bf16) trick:
//   x = hi + lo,  A*B ~= Ahi*Bhi + Ahi*Blo + Alo*Bhi   (fp32 accumulate)
// Tiles stream global->LDS via GLOBAL_LOAD_ASYNC_TO_LDS (ASYNCcnt) with a
// double-buffered LDS pipeline so loads overlap the WMMA stream.
// ---------------------------------------------------------------------------

typedef __attribute__((ext_vector_type(16))) __bf16 v16bf;
typedef __attribute__((ext_vector_type(8)))  __bf16 v8bf;
typedef __attribute__((ext_vector_type(4)))  __bf16 v4bf;
typedef __attribute__((ext_vector_type(8)))  float  v8f;
typedef __attribute__((ext_vector_type(4)))  int    i32x4;

#if defined(__HIP_DEVICE_COMPILE__) && \
    __has_builtin(__builtin_amdgcn_global_load_async_to_lds_b128)
#define USE_ASYNC_LDS 1
#else
#define USE_ASYNC_LDS 0
#endif

static __device__ __forceinline__ void cp16_g2l(__bf16* dst, const __bf16* src) {
#if USE_ASYNC_LDS
    __builtin_amdgcn_global_load_async_to_lds_b128(
        (__attribute__((address_space(1))) i32x4*)src,
        (__attribute__((address_space(3))) i32x4*)dst,
        0, 0);
#else
    *(v8bf*)dst = *(const v8bf*)src;
#endif
}

static __device__ __forceinline__ void async_copies_wait() {
#if USE_ASYNC_LDS
#if __has_builtin(__builtin_amdgcn_s_wait_asynccnt)
    __builtin_amdgcn_s_wait_asynccnt(0);
#else
    asm volatile("s_wait_asynccnt 0x0" ::: "memory");
#endif
#endif
}

static __device__ __forceinline__ v16bf mk16(v8bf a, v8bf b) {
    return __builtin_shufflevector(a, b, 0, 1, 2, 3, 4, 5, 6, 7,
                                         8, 9, 10, 11, 12, 13, 14, 15);
}

static __device__ __forceinline__ v8f wmma_bf16(v16bf a, v16bf b, v8f c) {
    // (neg_a, A, neg_b, B, c_mod, C, reuse_a, reuse_b)
    return __builtin_amdgcn_wmma_f32_16x16x32_bf16(false, a, false, b,
                                                   (short)0, c, false, false);
}

static __device__ __forceinline__ __bf16 bhi(float x) {
    return (__bf16)x;
}
static __device__ __forceinline__ __bf16 blo(float x) {
    __bf16 t = (__bf16)x;
    return (__bf16)(x - (float)t);
}

// ---------------------------------------------------------------------------
// Kernel 1: split a fp32 weight matrix into hi/lo bf16 planes (elementwise).
// ---------------------------------------------------------------------------
__global__ __launch_bounds__(256) void wsplit_kernel(
    const float* __restrict__ W, __bf16* __restrict__ Whi,
    __bf16* __restrict__ Wlo, int n4)
{
    int i = blockIdx.x * 256 + threadIdx.x;
    if (i >= n4) return;
    float4 w = ((const float4*)W)[i];
    v4bf hv, lv;
    hv[0] = bhi(w.x); lv[0] = blo(w.x);
    hv[1] = bhi(w.y); lv[1] = blo(w.y);
    hv[2] = bhi(w.z); lv[2] = blo(w.z);
    hv[3] = bhi(w.w); lv[3] = blo(w.w);
    ((v4bf*)Whi)[i] = hv;
    ((v4bf*)Wlo)[i] = lv;
}

// ---------------------------------------------------------------------------
// Kernel 2: per-row L2 normalize (dim=1) and split into hi/lo bf16 planes.
// One 256-thread block per row; D = 4096 -> 16 floats per thread.
// ---------------------------------------------------------------------------
__global__ __launch_bounds__(256) void normsplit_kernel(
    const float* __restrict__ H, __bf16* __restrict__ Ahi,
    __bf16* __restrict__ Alo, int D)
{
    const int row = blockIdx.x;
    const int tid = threadIdx.x;
    const float* hr = H + (size_t)row * D;

    float4 v[4];
    float s = 0.f;
#pragma unroll
    for (int i = 0; i < 4; ++i) {
        v[i] = ((const float4*)hr)[i * 256 + tid];
        s += v[i].x * v[i].x + v[i].y * v[i].y + v[i].z * v[i].z + v[i].w * v[i].w;
    }

    __shared__ float red[256];
    red[tid] = s;
    __syncthreads();
#pragma unroll
    for (int off = 128; off > 0; off >>= 1) {
        if (tid < off) red[tid] += red[tid + off];
        __syncthreads();
    }
    const float scale = 1.f / fmaxf(sqrtf(red[0]), 1e-12f);

    v4bf* hdst = (v4bf*)(Ahi + (size_t)row * D);
    v4bf* ldst = (v4bf*)(Alo + (size_t)row * D);
#pragma unroll
    for (int i = 0; i < 4; ++i) {
        float y0 = v[i].x * scale, y1 = v[i].y * scale;
        float y2 = v[i].z * scale, y3 = v[i].w * scale;
        v4bf hv, lv;
        hv[0] = bhi(y0); lv[0] = blo(y0);
        hv[1] = bhi(y1); lv[1] = blo(y1);
        hv[2] = bhi(y2); lv[2] = blo(y2);
        hv[3] = bhi(y3); lv[3] = blo(y3);
        hdst[i * 256 + tid] = hv;
        ldst[i * 256 + tid] = lv;
    }
}

// ---------------------------------------------------------------------------
// Kernel 3: C = relu(A @ W^T + bias) with A,W given as hi/lo bf16 planes.
// Workgroup tile 128x128, BK=32, 8 waves (4x2), each wave: 2x4 16x16 tiles.
// Double-buffered LDS, async global->LDS copies, 24 WMMA per wave per K-step.
// ---------------------------------------------------------------------------
__global__ __launch_bounds__(256, 1) void gemm_l2mlp_kernel(
    const __bf16* __restrict__ Ahi, const __bf16* __restrict__ Alo,
    const __bf16* __restrict__ Whi, const __bf16* __restrict__ Wlo,
    const float* __restrict__ bias, float* __restrict__ Cout,
    int D, int O)
{
    constexpr int BM = 128, BN = 128, BK = 32;
    constexpr int LDT = 40;  // padded LDS row stride (bf16): 80B -> bank-clean

    __shared__ __align__(16) __bf16 sAhi[2][BM * LDT];
    __shared__ __align__(16) __bf16 sAlo[2][BM * LDT];
    __shared__ __align__(16) __bf16 sWhi[2][BN * LDT];
    __shared__ __align__(16) __bf16 sWlo[2][BN * LDT];

    const int tid  = threadIdx.x;
    const int lane = tid & 31;
    const int wave = tid >> 5;
    const int wm = wave & 3;   // 4 waves span 128 rows (32 each)
    const int wn = wave >> 2;  // 2 waves span 128 cols (64 each)
    const int hh = lane >> 4;  // lane half
    const int rr = lane & 15;

    const size_t tileM = (size_t)blockIdx.y * BM;
    const size_t tileN = (size_t)blockIdx.x * BN;

    // cooperative tile load: thread -> (row, 16-elem k chunk)
    const int lrow = tid >> 1;
    const int lk   = (tid & 1) << 4;
    const size_t aoff = (tileM + lrow) * (size_t)D + lk;
    const size_t woff = (tileN + lrow) * (size_t)D + lk;
    const int    soff = lrow * LDT + lk;

    // issue the 8 x 16B copies that stage one K-step of all four tiles
    auto issue_stage = [&](int st, int kk) {
        cp16_g2l(&sAhi[st][soff],     Ahi + aoff + kk);
        cp16_g2l(&sAhi[st][soff + 8], Ahi + aoff + kk + 8);
        cp16_g2l(&sAlo[st][soff],     Alo + aoff + kk);
        cp16_g2l(&sAlo[st][soff + 8], Alo + aoff + kk + 8);
        cp16_g2l(&sWhi[st][soff],     Whi + woff + kk);
        cp16_g2l(&sWhi[st][soff + 8], Whi + woff + kk + 8);
        cp16_g2l(&sWlo[st][soff],     Wlo + woff + kk);
        cp16_g2l(&sWlo[st][soff + 8], Wlo + woff + kk + 8);
    };

    v8f acc[2][4] = {};

    issue_stage(0, 0);  // prologue

    int stage = 0;
    for (int k0 = 0; k0 < D; k0 += BK, stage ^= 1) {
        async_copies_wait();   // this wave's copies for `stage` have landed
        __syncthreads();       // all waves' copies landed; prev compute done
        if (k0 + BK < D)
            issue_stage(stage ^ 1, k0 + BK);  // overlaps with compute below

        const __bf16* cAhi = sAhi[stage];
        const __bf16* cAlo = sAlo[stage];
        const __bf16* cWhi = sWhi[stage];
        const __bf16* cWlo = sWlo[stage];

        // A fragments (16 contiguous k at 16h, matching B)
        v16bf fahi[2], falo[2];
#pragma unroll
        for (int mi = 0; mi < 2; ++mi) {
            const int ar = (wm * 32 + mi * 16 + rr) * LDT;
            fahi[mi] = mk16(*(const v8bf*)(cAhi + ar + 16 * hh),
                            *(const v8bf*)(cAhi + ar + 16 * hh + 8));
            falo[mi] = mk16(*(const v8bf*)(cAlo + ar + 16 * hh),
                            *(const v8bf*)(cAlo + ar + 16 * hh + 8));
        }

#pragma unroll
        for (int ni = 0; ni < 4; ++ni) {
            // B fragment: lane rr -> output col (W row), 16 contiguous k at 16h
            const int wr = (wn * 64 + ni * 16 + rr) * LDT + 16 * hh;
            v16bf fbhi = mk16(*(const v8bf*)(cWhi + wr),
                              *(const v8bf*)(cWhi + wr + 8));
            v16bf fblo = mk16(*(const v8bf*)(cWlo + wr),
                              *(const v8bf*)(cWlo + wr + 8));
#pragma unroll
            for (int mi = 0; mi < 2; ++mi) {
                acc[mi][ni] = wmma_bf16(fahi[mi], fbhi, acc[mi][ni]);
                acc[mi][ni] = wmma_bf16(fahi[mi], fblo, acc[mi][ni]);
                acc[mi][ni] = wmma_bf16(falo[mi], fbhi, acc[mi][ni]);
            }
        }
        __syncthreads();  // all waves done reading `stage` before next overwrite
    }

    // epilogue: bias + ReLU, fp32 store (D-matrix layout: VGPR j -> M=j+8h)
#pragma unroll
    for (int ni = 0; ni < 4; ++ni) {
        const size_t col = tileN + wn * 64 + ni * 16 + rr;
        const float bv = bias[col];
#pragma unroll
        for (int mi = 0; mi < 2; ++mi) {
#pragma unroll
            for (int j = 0; j < 8; ++j) {
                const size_t rowi = tileM + wm * 32 + mi * 16 + hh * 8 + j;
                float v = acc[mi][ni][j] + bv;
                Cout[rowi * (size_t)O + col] = v > 0.f ? v : 0.f;
            }
        }
    }
}

// ---------------------------------------------------------------------------
extern "C" void kernel_launch(void* const* d_in, const int* in_sizes, int n_in,
                              void* d_out, int out_size, void* d_ws, size_t ws_size,
                              hipStream_t stream) {
    (void)n_in; (void)out_size; (void)ws_size;
    const float* x = (const float*)d_in[0];
    const float* Ws[3] = {(const float*)d_in[1], (const float*)d_in[3], (const float*)d_in[5]};
    const float* bs[3] = {(const float*)d_in[2], (const float*)d_in[4], (const float*)d_in[6]};

    const int D = in_sizes[2];             // bias length = 4096
    const int Brows = in_sizes[0] / D;     // 8192
    const size_t BD = (size_t)Brows * D;
    const size_t DD = (size_t)D * D;

    __bf16* Ahi = (__bf16*)d_ws;
    __bf16* Alo = Ahi + BD;
    __bf16* Whi = Alo + BD;
    __bf16* Wlo = Whi + DD;
    float*  out = (float*)d_out;

    const int n4 = (int)(DD / 4);
    const dim3 ggrid(D / 128, Brows / 128);

    for (int l = 0; l < 3; ++l) {
        const float* hin = (l == 0) ? x : out + (size_t)(l - 1) * BD;
        wsplit_kernel<<<(n4 + 255) / 256, 256, 0, stream>>>(Ws[l], Whi, Wlo, n4);
        normsplit_kernel<<<Brows, 256, 0, stream>>>(hin, Ahi, Alo, D);
        gemm_l2mlp_kernel<<<ggrid, 256, 0, stream>>>(Ahi, Alo, Whi, Wlo, bs[l],
                                                     out + (size_t)l * BD, D, D);
    }
}
